// MultiHeadedAttention_54511724921122
// MI455X (gfx1250) — compile-verified, matrix operand order checked by reading, not run on hardware
//
#include <hip/hip_runtime.h>
#include <cmath>

// Problem constants (match reference)
#define BQ 4
#define TT 2048
#define CC 2048

typedef __bf16 bf16;
typedef __attribute__((ext_vector_type(4)))  __bf16 v4bf;
typedef __attribute__((ext_vector_type(8)))  __bf16 v8bf;
typedef __attribute__((ext_vector_type(16))) __bf16 v16bf;
typedef __attribute__((ext_vector_type(8)))  float  v8f;

// ---------------------------------------------------------------- convert
__global__ void cvt_f32_bf16_x4(const float* __restrict__ in,
                                bf16* __restrict__ out, int n4) {
    int i = blockIdx.x * blockDim.x + threadIdx.x;
    if (i < n4) {
        float4 v = ((const float4*)in)[i];
        v4bf b;
        b[0] = (bf16)v.x; b[1] = (bf16)v.y; b[2] = (bf16)v.z; b[3] = (bf16)v.w;
        ((v4bf*)out)[i] = b;
    }
}

// ---------------------------------------------------------------- GEMM
// Y[m,n] = scale * sum_k A[m,k] * B[n,k]   (both operands K-contiguous, ld = Kd)
// flags: bit0 -> bf16 output, bit1 -> causal mask (col>row => -1e30 pre-softmax),
//        bit2 -> transposed bf16 store (for V: out[b][n][s], m=b*TT+s)
#define F_OUTBF16 1
#define F_CAUSAL  2
#define F_TRANSV  4

#define TM 128
#define TN 128
#define TK 64
#define LDSK (TK + 8)   // row stride 72 bf16 = 144B (16B aligned)

// CDNA5 async global->LDS (ASYNCcnt-tracked, no VGPR round trip).
// GVS addressing: saddr = uniform 64-bit base, vaddr = per-lane 32-bit byte offset.
__device__ __forceinline__ void async_ld_b128(const void* lds_dst,
                                              const bf16* base,
                                              unsigned byte_off) {
    unsigned lds_addr = (unsigned)(size_t)lds_dst;   // low 32 bits = LDS offset
    asm volatile("global_load_async_to_lds_b128 %0, %1, %2"
                 :: "v"(lds_addr), "v"(byte_off), "s"(base)
                 : "memory");
}
__device__ __forceinline__ void wait_async0() {
    asm volatile("s_wait_asynccnt 0" ::: "memory");
}

__device__ __forceinline__ v16bf ld_frag(const bf16* p) {
    // A/B 16-bit fragment: lane holds K runs {8h..8h+7} and {16+8h..16+8h+7}
    v8bf lo = *(const v8bf*)p;
    v8bf hi = *(const v8bf*)(p + 16);
    return __builtin_shufflevector(lo, hi,
        0,1,2,3,4,5,6,7,8,9,10,11,12,13,14,15);
}

__global__ __launch_bounds__(256)
void gemm_bf16_wmma(const bf16* __restrict__ A, const bf16* __restrict__ Bm,
                    float* __restrict__ outF, bf16* __restrict__ outB,
                    int M, int N, int Kd, float scale, int flags) {
    __shared__ bf16 As[2][TM][LDSK];
    __shared__ bf16 Bs[2][TN][LDSK];

    const int tid  = threadIdx.x;
    const int lane = tid & 31;
    const int wave = tid >> 5;     // 0..7
    const int wm   = wave >> 1;    // 0..3 -> 32 rows each
    const int wn   = wave & 1;     // 0..1 -> 64 cols each
    const int lm   = lane & 15;    // row / col within 16x16 tile
    const int lh   = lane >> 4;    // K-half select

    const int n0 = blockIdx.x * TN;
    const int m0 = blockIdx.y * TM;

    v8f acc[2][4];
    const v8f vzero = {0.f,0.f,0.f,0.f,0.f,0.f,0.f,0.f};
#pragma unroll
    for (int mt = 0; mt < 2; ++mt)
#pragma unroll
        for (int nt = 0; nt < 4; ++nt) acc[mt][nt] = vzero;

    // cooperative async stage: 128x64 bf16 per tile = 256 thr * 4 chunks * 16B
    auto stage = [&](int buf, int kk) {
#pragma unroll
        for (int i = 0; i < 4; ++i) {
            int c  = tid + i * 256;        // 0..1023
            int r  = c >> 3;               // 0..127
            int c8 = (c & 7) * 8;
            async_ld_b128(&As[buf][r][c8], A,
                          (unsigned)(((size_t)(m0 + r) * Kd + kk + c8) * 2));
            async_ld_b128(&Bs[buf][r][c8], Bm,
                          (unsigned)(((size_t)(n0 + r) * Kd + kk + c8) * 2));
        }
    };

    stage(0, 0);
    int buf = 0;
    for (int kk = 0; kk < Kd; kk += TK) {
        wait_async0();        // this wave's async loads into `buf` have landed
        __syncthreads();      // ...and everyone else's too; prev buf reads done
        if (kk + TK < Kd) stage(buf ^ 1, kk + TK);   // prefetch next stage

#pragma unroll
        for (int ks = 0; ks < 2; ++ks) {
            const int kb = ks * 32;
            v16bf afrag[2], bfrag[4];
#pragma unroll
            for (int mt = 0; mt < 2; ++mt)
                afrag[mt] = ld_frag(&As[buf][wm * 32 + mt * 16 + lm][kb + lh * 8]);
#pragma unroll
            for (int nt = 0; nt < 4; ++nt)
                bfrag[nt] = ld_frag(&Bs[buf][wn * 64 + nt * 16 + lm][kb + lh * 8]);
#pragma unroll
            for (int mt = 0; mt < 2; ++mt)
#pragma unroll
                for (int nt = 0; nt < 4; ++nt)
                    acc[mt][nt] = __builtin_amdgcn_wmma_f32_16x16x32_bf16(
                        false, afrag[mt], false, bfrag[nt],
                        (short)0, acc[mt][nt], false, false);
        }
        buf ^= 1;
    }

    // epilogue: C/D layout: lane col = lm, VGPR r -> row = 8*lh + r
#pragma unroll
    for (int mt = 0; mt < 2; ++mt) {
#pragma unroll
        for (int nt = 0; nt < 4; ++nt) {
            const int gr0 = m0 + wm * 32 + mt * 16 + lh * 8;
            const int gc  = n0 + wn * 64 + nt * 16 + lm;
#pragma unroll
            for (int r = 0; r < 8; ++r) {
                const int grow = gr0 + r;
                float v = acc[mt][nt][r] * scale;
                if ((flags & F_CAUSAL) && (gc > grow)) v = -1.0e30f;
                if (flags & F_TRANSV) {
                    int b = grow / TT, s = grow % TT;
                    outB[(size_t)b * CC * TT + (size_t)gc * TT + s] = (bf16)v;
                } else if (flags & F_OUTBF16) {
                    outB[(size_t)grow * N + gc] = (bf16)v;
                } else {
                    outF[(size_t)grow * N + gc] = v;
                }
            }
        }
    }
}

// ---------------------------------------------------------------- softmax
__global__ __launch_bounds__(256)
void row_softmax_to_bf16(const float* __restrict__ S, bf16* __restrict__ P) {
    __shared__ float red[256];
    const int row = blockIdx.x;
    const float* s = S + (size_t)row * TT;

    float mx = -3.0e38f;
    for (int c = threadIdx.x; c < TT; c += 256) mx = fmaxf(mx, s[c]);
    red[threadIdx.x] = mx;
    __syncthreads();
    for (int off = 128; off > 0; off >>= 1) {
        if (threadIdx.x < off)
            red[threadIdx.x] = fmaxf(red[threadIdx.x], red[threadIdx.x + off]);
        __syncthreads();
    }
    mx = red[0];
    __syncthreads();

    float sum = 0.f;
    for (int c = threadIdx.x; c < TT; c += 256) sum += __expf(s[c] - mx);
    red[threadIdx.x] = sum;
    __syncthreads();
    for (int off = 128; off > 0; off >>= 1) {
        if (threadIdx.x < off)
            red[threadIdx.x] += red[threadIdx.x + off];
        __syncthreads();
    }
    const float inv = 1.0f / red[0];

    for (int c = threadIdx.x; c < TT; c += 256)
        P[(size_t)row * TT + c] = (bf16)(__expf(s[c] - mx) * inv);
}

// ---------------------------------------------------------------- launch
extern "C" void kernel_launch(void* const* d_in, const int* in_sizes, int n_in,
                              void* d_out, int out_size, void* d_ws, size_t ws_size,
                              hipStream_t stream) {
    (void)in_sizes; (void)n_in; (void)out_size; (void)ws_size;

    const float* x  = (const float*)d_in[0];
    const float* Wq = (const float*)d_in[1];
    const float* Wk = (const float*)d_in[2];
    const float* Wv = (const float*)d_in[3];
    const float* Wo = (const float*)d_in[4];
    float* out = (float*)d_out;

    const int M   = BQ * TT;                 // 8192
    const size_t nX = (size_t)BQ * TT * CC;  // 16,777,216
    const size_t nW = (size_t)CC * CC;       // 4,194,304

    char* ws = (char*)d_ws;
    size_t off = 0;
    bf16* xb  = (bf16*)(ws + off); off += nX * 2;            //  32 MB
    bf16* Wqb = (bf16*)(ws + off); off += nW * 2;            //   8 MB
    bf16* Wkb = (bf16*)(ws + off); off += nW * 2;
    bf16* Wvb = (bf16*)(ws + off); off += nW * 2;
    bf16* Wob = (bf16*)(ws + off); off += nW * 2;
    bf16* Qb  = (bf16*)(ws + off); off += nX * 2;            //  32 MB
    bf16* Kb  = (bf16*)(ws + off); off += nX * 2;
    bf16* Vt  = (bf16*)(ws + off); off += nX * 2;            //  V^T per batch [C][T]
    bf16* Ob  = (bf16*)(ws + off); off += nX * 2;
    float* Sf = (float*)(ws + off); off += (size_t)TT * TT * 4;  // 16 MB (reused per batch)
    bf16*  Pb = (bf16*)(ws + off);  off += (size_t)TT * TT * 2;  //  8 MB (reused per batch)

    // 1) convert inputs to bf16
    cvt_f32_bf16_x4<<<(int)(nX / 4 / 256), 256, 0, stream>>>(x,  xb,  (int)(nX / 4));
    cvt_f32_bf16_x4<<<(int)(nW / 4 / 256), 256, 0, stream>>>(Wq, Wqb, (int)(nW / 4));
    cvt_f32_bf16_x4<<<(int)(nW / 4 / 256), 256, 0, stream>>>(Wk, Wkb, (int)(nW / 4));
    cvt_f32_bf16_x4<<<(int)(nW / 4 / 256), 256, 0, stream>>>(Wv, Wvb, (int)(nW / 4));
    cvt_f32_bf16_x4<<<(int)(nW / 4 / 256), 256, 0, stream>>>(Wo, Wob, (int)(nW / 4));

    const dim3 blk(256);
    const dim3 gProj(CC / TN, M / TM);   // 16 x 64
    const dim3 gAttn(TT / TN, TT / TM);  // 16 x 16
    const dim3 gPV(CC / TN, TT / TM);    // 16 x 16

    // 2) projections: Q,K row-major bf16; V stored transposed per batch
    gemm_bf16_wmma<<<gProj, blk, 0, stream>>>(xb, Wqb, nullptr, Qb, M, CC, CC, 1.0f, F_OUTBF16);
    gemm_bf16_wmma<<<gProj, blk, 0, stream>>>(xb, Wkb, nullptr, Kb, M, CC, CC, 1.0f, F_OUTBF16);
    gemm_bf16_wmma<<<gProj, blk, 0, stream>>>(xb, Wvb, nullptr, Vt, M, CC, CC, 1.0f, F_OUTBF16 | F_TRANSV);

    // 3) attention per batch (S and P buffers reused; stream-serialized)
    const float rscale = 1.0f / sqrtf((float)CC);
    for (int b = 0; b < BQ; ++b) {
        const size_t boff = (size_t)b * TT * CC;
        // S = scale * Q K^T with causal mask -> fp32
        gemm_bf16_wmma<<<gAttn, blk, 0, stream>>>(Qb + boff, Kb + boff,
                                                  Sf, nullptr, TT, TT, CC, rscale, F_CAUSAL);
        // P = softmax(S) -> bf16
        row_softmax_to_bf16<<<TT, blk, 0, stream>>>(Sf, Pb);
        // O = P V  (V^T is K-contiguous) -> bf16
        gemm_bf16_wmma<<<gPV, blk, 0, stream>>>(Pb, Vt + boff,
                                                nullptr, Ob + boff, TT, CC, TT, 1.0f, F_OUTBF16);
    }

    // 4) output projection -> fp32 d_out
    gemm_bf16_wmma<<<gProj, blk, 0, stream>>>(Ob, Wob, out, nullptr, M, CC, CC, 1.0f, 0);
}
